// PointTransformerV3_43843026157917
// MI455X (gfx1250) — matrix operand order repaired and checked by reading, not compiled, not run on hardware
//
#include <hip/hip_runtime.h>
#include <hip/hip_bf16.h>

typedef unsigned int u32;
typedef unsigned short u16;
typedef unsigned long long u64;
typedef __attribute__((ext_vector_type(4)))  float        v4f;
typedef __attribute__((ext_vector_type(4)))  unsigned int v4u;
typedef __attribute__((ext_vector_type(8)))  float        v8f;
typedef __attribute__((ext_vector_type(16))) __bf16       v16bf;

#define NPTS   131072
#define CCH    512
#define NH     8
#define KPATCH 128
#define HDIM   64
#define NPAT   1024      // NPTS / KPATCH
#define RPE_N  33
#define POSB   16

// ---------- bf16 helpers (round-to-nearest-even) ----------
__device__ inline u16 bfbits(float f) {
    u32 u = __float_as_uint(f);
    u32 r = u + 0x7FFFu + ((u >> 16) & 1u);
    return (u16)(r >> 16);
}
__device__ inline __bf16 f2bf(float f) {
    union { u16 s; __bf16 b; } cv; cv.s = bfbits(f); return cv.b;
}
__device__ inline u32 pack2(float a, float b) {
    return (u32)bfbits(a) | ((u32)bfbits(b) << 16);
}

// ---------- CDNA5 async Global->LDS copy (ASYNCcnt path) ----------
// GVS mode: mem = SGPR64 base + VGPR32 offset; LDS dest = per-lane VGPR byte address.
__device__ inline void async_copy_b128(u32 lds_addr, const void* gbase, u32 voff) {
    asm volatile("global_load_async_to_lds_b128 %0, %1, %2"
                 :: "v"(lds_addr), "v"(voff), "s"((u64)(uintptr_t)gbase)
                 : "memory");
}
__device__ inline void wait_async() {
    asm volatile("s_wait_asynccnt 0x0" ::: "memory");
}
__device__ inline u32 lds_off(const void* p) {   // low 32 bits of generic ptr = LDS byte addr
    return (u32)(uintptr_t)p;
}

// ---------- WMMA fragment loaders ----------
// A-matrix 16x32 bf16 (ISA 7.12.2): lane L -> M=L%16; low lanes: K dwords {0..3, 8..11},
// high lanes: K dwords {4..7, 12..15}.  ks = K offset (0 or 32 within a 64-wide stage).
__device__ inline v16bf loadFragA(const __bf16* base, int row, int stride, int ks, int laneHi) {
    const u32* p = (const u32*)(base + row * stride) + (ks >> 1) + laneHi * 4;
    union { v4u q[2]; v16bf v; } f;
    f.q[0] = *(const v4u*)(p);
    f.q[1] = *(const v4u*)(p + 8);
    return f.v;
}
// B-matrix 32x16 bf16: lane L -> N=L%16; low lanes: K dwords {0..3, 8..11}, high lanes: K dwords {4..7, 12..15}.
__device__ inline v16bf loadFragB(const __bf16* base, int row, int stride, int ks, int laneHi) {
    const u32* p = (const u32*)(base + row * stride) + (ks >> 1) + laneHi * 4;
    union { v4u q[2]; v16bf v; } f;
    f.q[0] = *(const v4u*)(p);
    f.q[1] = *(const v4u*)(p + 8);
    return f.v;
}
__device__ inline v8f wmma_bf16(v16bf a, v16bf b, v8f c) {
    return __builtin_amdgcn_wmma_f32_16x16x32_bf16(false, a, false, b, (short)0, c, false, false);
}

// ---------- Kernel 1: weight fp32 -> bf16 ----------
__global__ __launch_bounds__(256) void conv_weights(const float* __restrict__ qkvw,
                                                    const float* __restrict__ projw,
                                                    __bf16* __restrict__ qkvw_bf,
                                                    __bf16* __restrict__ projw_bf) {
    int id = blockIdx.x * 256 + threadIdx.x;
    const int nq = 3 * CCH * CCH;      // 786432
    const int np = CCH * CCH;          // 262144
    if (id < nq)            qkvw_bf[id]      = f2bf(qkvw[id]);
    else if (id < nq + np)  projw_bf[id - nq] = f2bf(projw[id - nq]);
}

// ---------- Kernel 2: gather + QKV GEMM (M=131072, N=1536, K=512) ----------
// Epilogue scatters: Q,K -> [p][h][kk][d] bf16 ; V -> [p][h][d][kk] bf16 (transposed)
__global__ __launch_bounds__(256) void qkv_gemm(const float* __restrict__ feat,
                                                const int* __restrict__ order,
                                                const __bf16* __restrict__ wbf,
                                                const float* __restrict__ bias,
                                                __bf16* __restrict__ Qd,
                                                __bf16* __restrict__ Kd,
                                                __bf16* __restrict__ Vd) {
    __shared__ __bf16 As[128 * 64];
    __shared__ __bf16 Bs[128 * 64];
    const int tid = threadIdx.x;
    const int bn = blockIdx.x;      // 0..11
    const int bm = blockIdx.y;      // 0..1023
    const int wave = tid >> 5, lane = tid & 31;
    const int laneLow = lane & 15, laneHi = lane >> 4;

    int grow[8], arow[8], acol[8];
#pragma unroll
    for (int u = 0; u < 8; u++) {
        int q = tid + u * 256;               // 0..2047
        arow[u] = q >> 4; acol[u] = q & 15;  // row 0..127, float4 col 0..15
        grow[u] = order[bm * 128 + arow[u]];
    }
    // B-stage async copy addressing (constant across kt except scalar base)
    u32 bvoff[4];
    const u32 ldsB = lds_off(Bs);
#pragma unroll
    for (int u = 0; u < 4; u++) {
        int q = tid + u * 256; int row = q >> 3, c = q & 7;
        bvoff[u] = (u32)(row * CCH + c * 8) * 2u;    // byte offset within weight row block
    }

    v8f acc[8];
#pragma unroll
    for (int t = 0; t < 8; t++)
#pragma unroll
        for (int e = 0; e < 8; e++) acc[t][e] = 0.0f;

    for (int kt = 0; kt < 8; kt++) {
        const int K0 = kt * 64;
        // stage B: bf16 weight rows via async Global->LDS DMA
        const void* bbase = (const void*)(wbf + (long)bn * 128 * CCH + K0);
#pragma unroll
        for (int u = 0; u < 4; u++) {
            int q = tid + u * 256;
            async_copy_b128(ldsB + (u32)q * 16u, bbase, bvoff[u]);
        }
        // stage A: gathered feat rows, fp32 -> bf16 (manual: conversion required)
#pragma unroll
        for (int u = 0; u < 8; u++) {
            v4f f4 = *(const v4f*)(feat + (long)grow[u] * CCH + K0 + acol[u] * 4);
            u32* dst = (u32*)As + arow[u] * 32 + acol[u] * 2;
            dst[0] = pack2(f4.x, f4.y);
            dst[1] = pack2(f4.z, f4.w);
        }
        wait_async();
        __syncthreads();
#pragma unroll
        for (int ks2 = 0; ks2 < 2; ks2++) {
            v16bf a = loadFragA(As, wave * 16 + laneLow, 64, ks2 * 32, laneHi);
#pragma unroll
            for (int nt = 0; nt < 8; nt++) {
                v16bf b = loadFragB(Bs, nt * 16 + laneLow, 64, ks2 * 32, laneHi);
                acc[nt] = wmma_bf16(a, b, acc[nt]);
            }
        }
        __syncthreads();
    }

    // epilogue: scatter into attention-friendly layouts
#pragma unroll
    for (int nt = 0; nt < 8; nt++) {
        int j = bn * 128 + nt * 16 + laneLow;        // 0..1535
        int which = j >> 9, h = (j >> 6) & 7, d = j & 63;
        float bj = bias[j];
#pragma unroll
        for (int r = 0; r < 8; r++) {
            int i = bm * 128 + wave * 16 + r + 8 * laneHi;
            int p = i >> 7, kk = i & 127;
            __bf16 bv = f2bf(acc[nt][r] + bj);
            long ph = (long)(p * NH + h);
            if (which == 0)      Qd[(ph * 128 + kk) * 64 + d] = bv;
            else if (which == 1) Kd[(ph * 128 + kk) * 64 + d] = bv;
            else                 Vd[(ph * 64 + d) * 128 + kk] = bv;   // transposed
        }
    }
}

// ---------- Kernel 3: attention per (patch, head) ----------
__global__ __launch_bounds__(256) void attn_kernel(const __bf16* __restrict__ Qd,
                                                   const __bf16* __restrict__ Kd,
                                                   const __bf16* __restrict__ Vd,
                                                   const int* __restrict__ order,
                                                   const int* __restrict__ gridc,
                                                   const float* __restrict__ rpe,
                                                   __bf16* __restrict__ AO) {
    __shared__ __bf16 Ks[128 * 64];   // key  rows [kk][d]
    __shared__ __bf16 Vs[64 * 128];   // V^T  rows [d][kk]
    __shared__ __bf16 Ps[128 * 64];   // softmax probs, half of K-dim at a time
    __shared__ int    Cs[128 * 3];
    __shared__ float  Rs[3 * RPE_N];  // 99

    const int tid = threadIdx.x;
    const int pb = blockIdx.x >> 3, h = blockIdx.x & 7;
    const long base = (long)(pb * NH + h) * (128 * 64);
    const int wave = tid >> 5, lane = tid & 31;
    const int laneLow = lane & 15, laneHi = lane >> 4;

    // K and V tiles: pure bf16 copies -> async Global->LDS DMA
    const u32 ldsK = lds_off(Ks), ldsV = lds_off(Vs);
    const void* kbase = (const void*)(Kd + base);
    const void* vbase = (const void*)(Vd + base);
#pragma unroll
    for (int u = 0; u < 4; u++) {
        u32 q = (u32)(tid + u * 256);
        async_copy_b128(ldsK + q * 16u, kbase, q * 16u);
        async_copy_b128(ldsV + q * 16u, vbase, q * 16u);
    }
    if (tid < 128) {
        int idx = order[pb * 128 + tid];
        Cs[tid * 3 + 0] = gridc[idx * 3 + 0];
        Cs[tid * 3 + 1] = gridc[idx * 3 + 1];
        Cs[tid * 3 + 2] = gridc[idx * 3 + 2];
    }
    if (tid < 3 * RPE_N) Rs[tid] = rpe[tid * NH + h];

    // Q fragments straight from global (A layout)
    v16bf qf0 = loadFragA(Qd + base, wave * 16 + laneLow, 64, 0, laneHi);
    v16bf qf1 = loadFragA(Qd + base, wave * 16 + laneLow, 64, 32, laneHi);

    wait_async();
    __syncthreads();

    // S = Q K^T
    v8f s[8];
#pragma unroll
    for (int t = 0; t < 8; t++)
#pragma unroll
        for (int e = 0; e < 8; e++) s[t][e] = 0.0f;
#pragma unroll
    for (int ks2 = 0; ks2 < 2; ks2++) {
        v16bf a = ks2 ? qf1 : qf0;
#pragma unroll
        for (int nt = 0; nt < 8; nt++) {
            v16bf b = loadFragB(Ks, nt * 16 + laneLow, 64, ks2 * 32, laneHi);
            s[nt] = wmma_bf16(a, b, s[nt]);
        }
    }

    // scale + RPE bias
    const float scale = 0.125f;   // HD^-0.5
#pragma unroll
    for (int nt = 0; nt < 8; nt++) {
        int j = nt * 16 + laneLow;
        int cjx = Cs[j * 3 + 0], cjy = Cs[j * 3 + 1], cjz = Cs[j * 3 + 2];
#pragma unroll
        for (int r = 0; r < 8; r++) {
            int i = wave * 16 + r + 8 * laneHi;
            int dx = Cs[i * 3 + 0] - cjx;
            int dy = Cs[i * 3 + 1] - cjy;
            int dz = Cs[i * 3 + 2] - cjz;
            dx = dx < -POSB ? -POSB : (dx > POSB ? POSB : dx);
            dy = dy < -POSB ? -POSB : (dy > POSB ? POSB : dy);
            dz = dz < -POSB ? -POSB : (dz > POSB ? POSB : dz);
            float bias = Rs[POSB + dx] + Rs[RPE_N + POSB + dy] + Rs[2 * RPE_N + POSB + dz];
            s[nt][r] = s[nt][r] * scale + bias;
        }
    }

    // softmax over j (row lives in one vgpr index r across 8 tiles x 16 lanes of a half)
#pragma unroll
    for (int r = 0; r < 8; r++) {
        float mx = s[0][r];
#pragma unroll
        for (int nt = 1; nt < 8; nt++) mx = fmaxf(mx, s[nt][r]);
        for (int off = 1; off < 16; off <<= 1) mx = fmaxf(mx, __shfl_xor(mx, off, 32));
        float sum = 0.0f;
#pragma unroll
        for (int nt = 0; nt < 8; nt++) { float e = __expf(s[nt][r] - mx); s[nt][r] = e; sum += e; }
        for (int off = 1; off < 16; off <<= 1) sum += __shfl_xor(sum, off, 32);
        float inv = 1.0f / sum;
#pragma unroll
        for (int nt = 0; nt < 8; nt++) s[nt][r] *= inv;
    }

    // O = P V  (process K-dim 128 in two 64-halves through per-wave Ps rows)
    v8f o[4];
#pragma unroll
    for (int t = 0; t < 4; t++)
#pragma unroll
        for (int e = 0; e < 8; e++) o[t][e] = 0.0f;

#pragma unroll
    for (int half = 0; half < 2; half++) {
#pragma unroll
        for (int nt2 = 0; nt2 < 4; nt2++) {
            int nt = half * 4 + nt2;
#pragma unroll
            for (int r = 0; r < 8; r++) {
                Ps[(wave * 16 + r + 8 * laneHi) * 64 + nt2 * 16 + laneLow] = f2bf(s[nt][r]);
            }
        }
        // each wave reads only its own 16 rows: same-wave DS ordering, no barrier needed
#pragma unroll
        for (int ks2 = 0; ks2 < 2; ks2++) {
            v16bf a = loadFragA(Ps, wave * 16 + laneLow, 64, ks2 * 32, laneHi);
#pragma unroll
            for (int dt = 0; dt < 4; dt++) {
                v16bf b = loadFragB(Vs, dt * 16 + laneLow, 128, half * 64 + ks2 * 32, laneHi);
                o[dt] = wmma_bf16(a, b, o[dt]);
            }
        }
    }

    // write attention out (serialized row-major N x 512, bf16)
#pragma unroll
    for (int dt = 0; dt < 4; dt++) {
#pragma unroll
        for (int r = 0; r < 8; r++) {
            int i = pb * 128 + wave * 16 + r + 8 * laneHi;
            int c = h * 64 + dt * 16 + laneLow;
            AO[(long)i * CCH + c] = f2bf(o[dt][r]);
        }
    }
}

// ---------- Kernel 4: projection GEMM + scatter (M=131072, N=512, K=512) ----------
__global__ __launch_bounds__(256) void proj_gemm(const __bf16* __restrict__ AO,
                                                 const __bf16* __restrict__ wbf,
                                                 const float* __restrict__ bias,
                                                 const int* __restrict__ order,
                                                 float* __restrict__ out) {
    __shared__ __bf16 As[128 * 64];
    __shared__ __bf16 Bs[128 * 64];
    const int tid = threadIdx.x;
    const int bn = blockIdx.x;      // 0..3
    const int bm = blockIdx.y;      // 0..1023
    const int wave = tid >> 5, lane = tid & 31;
    const int laneLow = lane & 15, laneHi = lane >> 4;

    const u32 ldsA = lds_off(As), ldsB = lds_off(Bs);
    u32 cvoff[4];
#pragma unroll
    for (int u = 0; u < 4; u++) {
        int q = tid + u * 256; int row = q >> 3, c = q & 7;
        cvoff[u] = (u32)(row * CCH + c * 8) * 2u;
    }

    v8f acc[8];
#pragma unroll
    for (int t = 0; t < 8; t++)
#pragma unroll
        for (int e = 0; e < 8; e++) acc[t][e] = 0.0f;

    for (int kt = 0; kt < 8; kt++) {
        const int K0 = kt * 64;
        const void* abase = (const void*)(AO + (long)bm * 128 * CCH + K0);
        const void* bbase = (const void*)(wbf + (long)bn * 128 * CCH + K0);
#pragma unroll
        for (int u = 0; u < 4; u++) {
            u32 q = (u32)(tid + u * 256);
            async_copy_b128(ldsA + q * 16u, abase, cvoff[u]);
            async_copy_b128(ldsB + q * 16u, bbase, cvoff[u]);
        }
        wait_async();
        __syncthreads();
#pragma unroll
        for (int ks2 = 0; ks2 < 2; ks2++) {
            v16bf a = loadFragA(As, wave * 16 + laneLow, 64, ks2 * 32, laneHi);
#pragma unroll
            for (int nt = 0; nt < 8; nt++) {
                v16bf b = loadFragB(Bs, nt * 16 + laneLow, 64, ks2 * 32, laneHi);
                acc[nt] = wmma_bf16(a, b, acc[nt]);
            }
        }
        __syncthreads();
    }

#pragma unroll
    for (int nt = 0; nt < 8; nt++) {
        int j = bn * 128 + nt * 16 + laneLow;
        float bj = bias[j];
#pragma unroll
        for (int r = 0; r < 8; r++) {
            int i = bm * 128 + wave * 16 + r + 8 * laneHi;
            out[(long)order[i] * CCH + j] = acc[nt][r] + bj;  // scatter back to input order
        }
    }
}

// ---------- launch ----------
extern "C" void kernel_launch(void* const* d_in, const int* in_sizes, int n_in,
                              void* d_out, int out_size, void* d_ws, size_t ws_size,
                              hipStream_t stream) {
    const float* feat  = (const float*)d_in[0];
    const int*   gridc = (const int*)d_in[1];
    const int*   order = (const int*)d_in[2];
    // d_in[3] = inverse (unused; we scatter via order)
    const float* qkvw  = (const float*)d_in[4];
    const float* qkvb  = (const float*)d_in[5];
    const float* projw = (const float*)d_in[6];
    const float* projb = (const float*)d_in[7];
    const float* rpe   = (const float*)d_in[8];
    float* out = (float*)d_out;

    char* ws = (char*)d_ws;
    __bf16* qkvw_bf = (__bf16*)(ws);                      // 786432 elems  (1,572,864 B)
    __bf16* projw_bf = (__bf16*)(ws + 1572864);           // 262144 elems  (524,288 B)
    const long TILEBYTES = (long)NPTS * CCH * 2;          // 134,217,728 B each
    __bf16* Qd = (__bf16*)(ws + 2097152);
    __bf16* Kd = (__bf16*)(ws + 2097152 + TILEBYTES);
    __bf16* Vd = (__bf16*)(ws + 2097152 + 2 * TILEBYTES);
    __bf16* AO = (__bf16*)(ws + 2097152 + 3 * TILEBYTES);

    conv_weights<<<4096, 256, 0, stream>>>(qkvw, projw, qkvw_bf, projw_bf);
    qkv_gemm<<<dim3(12, 1024), 256, 0, stream>>>(feat, order, qkvw_bf, qkvb, Qd, Kd, Vd);
    attn_kernel<<<NPAT * NH, 256, 0, stream>>>(Qd, Kd, Vd, order, gridc, rpe, AO);
    proj_gemm<<<dim3(4, 1024), 256, 0, stream>>>(AO, projw_bf, projb, order, out);
}
